// TerrainCompletionNetSparse_45114336477868
// MI455X (gfx1250) — matrix operand order repaired and checked by reading, not compiled, not run on hardware
//
#include <hip/hip_runtime.h>
#include <stdint.h>

// ---------------------------------------------------------------------------
// CDNA5 (gfx1250) TerrainCompletionNetSparse dense analog.
// Channel-last bf16 activations, WMMA f32_16x16x32_bf16 implicit-GEMM convs,
// fragment-ordered packed weights staged via double-buffered
// global_load_async_to_lds, fused instance-norm stats in the GEMM epilogue.
// All spatial extents are powers of two -> shift/mask addressing throughout.
// ---------------------------------------------------------------------------

typedef __attribute__((ext_vector_type(16))) __bf16 bf16x16;
typedef __attribute__((ext_vector_type(8)))  float  floatx8;

union AFrag {
  uint4    q[2];
  uint32_t u[8];
  bf16x16  v;
};

struct ConvParams {
  const __bf16* act;   // [N][IH][IW][ID][IT][Cin]   channel-last bf16
  const __bf16* wpk;   // fragment-ordered packed weights (see pack kernel)
  float* out;          // [N][Cout][nvox] fp32
  float* stats;        // [N][Cout][2] (sum, sumsq) via atomics
  int Cin, Cout;
  int IH, IW, ID, IT;
  int OH, OW, OD, OT;
  int OTb, ODb, OWb;   // log2(OT), log2(OD), log2(OW)
  int KH, KW, KD, KT;
  int sh, ph, pt;      // isotropic spatial stride/pad; T pad
  int transpose;
  int nCC;             // ceil(Cin/32)
  int nCoTt;           // 2*ceil(Cout/32)  (16-channel tiles, padded to pairs)
};

// One output-coordinate -> input-coordinate map per dimension.
__device__ __forceinline__ void map_dim(int o, int k, int I, int s, int p,
                                        int tr, int& idx, bool& ok) {
  if (tr) {  // gather form of stride-2 transpose conv
    int t = o + p - k;
    ok = (t >= 0) && ((t & 1) == 0) && ((t >> 1) < I);
    idx = ok ? (t >> 1) : 0;
  } else {
    int t = o * s + k - p;
    ok = (t >= 0) && (t < I);
    idx = ok ? t : 0;
  }
}

// Stage one 2 KB chunk pair (two 16-channel B tiles) into LDS.
// INST_OFFSET is added to BOTH the LDS and global address, so one base
// register pair serves all four 512 B slices.
__device__ __forceinline__ void stage_chunk(uint32_t ldsBase, const __bf16* g,
                                            int lane) {
  uint64_t ga = (uint64_t)(uintptr_t)(g + (size_t)lane * 32);
  uint32_t la = ldsBase + (uint32_t)lane * 64u;
  asm volatile("global_load_async_to_lds_b128 %0, %1, off"           :: "v"(la), "v"(ga) : "memory");
  asm volatile("global_load_async_to_lds_b128 %0, %1, off offset:16" :: "v"(la), "v"(ga) : "memory");
  asm volatile("global_load_async_to_lds_b128 %0, %1, off offset:32" :: "v"(la), "v"(ga) : "memory");
  asm volatile("global_load_async_to_lds_b128 %0, %1, off offset:48" :: "v"(la), "v"(ga) : "memory");
}

// Implicit-GEMM conv. Block = 256 threads = 8 wave32. Each wave owns a
// 16-voxel M tile; the block owns a 32-channel N tile (2 WMMA accumulators
// sharing each A fragment). B chunks are double-buffered in LDS via async DMA.
__global__ __launch_bounds__(256) void tcn_conv3d_wmma_bf16(ConvParams P) {
  __shared__ bf16x16 sB[2][64];        // 2 x (2 tiles x 32 lanes x 32B) = 4 KB
  __shared__ float sSum[32], sSq[32];
  const int tid  = threadIdx.x;
  const int lane = tid & 31;
  const int wave = tid >> 5;
  const int half = lane >> 4;          // wave32 half per WMMA VGPR layouts
  const int nl   = lane & 15;
  const int n  = blockIdx.z;
  const int nt = blockIdx.y;           // 32-channel tile
  const long nvox  = (long)P.OH * P.OW * P.OD * P.OT;
  const long mRow  = (long)blockIdx.x * 128 + wave * 16 + nl;   // A-row voxel
  const bool mvalid = mRow < nvox;
  long mm = mvalid ? mRow : 0;
  const int ot = (int)(mm & (P.OT - 1)); long r0 = mm >> P.OTb;
  const int od = (int)(r0 & (P.OD - 1)); r0 >>= P.ODb;
  const int ow = (int)(r0 & (P.OW - 1));
  const int oh = (int)(r0 >> P.OWb);

  const uint32_t lds0 = (uint32_t)(uintptr_t)(&sB[0][0]);
  floatx8 acc0 = {}, acc1 = {};

  const __bf16* gB = P.wpk + (size_t)nt * 1024;   // ct pair for this tile
  const size_t gStride = (size_t)P.nCoTt * 512;   // next (kpos,cc) chunk pair
  const int totalChunks = P.KH * P.KW * P.KD * P.KT * P.nCC;
  if (wave == 0) stage_chunk(lds0, gB, lane);
  int chunkCtr = 0, buf = 0;

  for (int kh = 0; kh < P.KH; ++kh) {
    int ih; bool okh; map_dim(oh, kh, P.IH, P.sh, P.ph, P.transpose, ih, okh);
    for (int kw = 0; kw < P.KW; ++kw) {
      int iw; bool okw; map_dim(ow, kw, P.IW, P.sh, P.ph, P.transpose, iw, okw);
      for (int kd = 0; kd < P.KD; ++kd) {
        int id; bool okd; map_dim(od, kd, P.ID, P.sh, P.ph, P.transpose, id, okd);
        for (int kt = 0; kt < P.KT; ++kt) {
          int it = ot + kt - P.pt;                 // T: stride 1 always
          bool okt = (it >= 0) && (it < P.IT);
          bool vOK = mvalid & okh & okw & okd & okt;
          size_t row = ((((size_t)n * P.IH + ih) * P.IW + iw) * P.ID + id) * P.IT + it;
          row = vOK ? row : 0;                     // safe dummy row
          const __bf16* rowPtr = P.act + row * (size_t)P.Cin;
          __builtin_prefetch(rowPtr, 0, 3);
          for (int cc = 0; cc < P.nCC; ++cc) {
            if (wave == 0) {
              if (chunkCtr + 1 < totalChunks) {
                stage_chunk(lds0 + (uint32_t)(buf ^ 1) * 2048u, gB + gStride, lane);
                asm volatile("s_wait_asynccnt 4" ::: "memory");  // chunk i done
              } else {
                asm volatile("s_wait_asynccnt 0" ::: "memory");
              }
            }
            __syncthreads();
            bf16x16 b0 = sB[buf][lane];        // ds_load_b128 x2: fragment image
            bf16x16 b1 = sB[buf][32 + lane];

            AFrag a;
            const int kRem = P.Cin - cc * 32;
            if (kRem >= 32) {
              // channel-last: each half-wave's K values are one contiguous 32 B run
              const uint4* ap4 = (const uint4*)(rowPtr + cc * 32);
              uint4 q0 = ap4[2 * half];
              uint4 q1 = ap4[2 * half + 1];
              a.q[0] = q0; a.q[1] = q1;
#pragma unroll
              for (int d = 0; d < 8; ++d) a.u[d] = vOK ? a.u[d] : 0u;
            } else {
              // small-Cin layers (3/8/16): clamped, branch-free masked gather
              const __bf16* ap = rowPtr + cc * 32;
#pragma unroll
              for (int e = 0; e < 16; ++e) {
                int k = half * 16 + e;
                int kc = k < kRem ? k : 0;
                __bf16 x = ap[kc];
                a.v[e] = (vOK && k < kRem) ? x : (__bf16)0.0f;
              }
            }
            acc0 = __builtin_amdgcn_wmma_f32_16x16x32_bf16(
                false, a.v, false, b0, (short)0, acc0, false, false);
            acc1 = __builtin_amdgcn_wmma_f32_16x16x32_bf16(
                false, a.v, false, b1, (short)0, acc1, false, false);
            __syncthreads();   // buf[cur] free for next-next stage
            gB += gStride; ++chunkCtr; buf ^= 1;
          }
        }
      }
    }
  }

  // Epilogue: fp32 store + fused instance-norm partial sums (LDS then global).
  if (tid < 32) { sSum[tid] = 0.f; sSq[tid] = 0.f; }
  __syncthreads();
  const long mBase = (long)blockIdx.x * 128 + wave * 16 + half * 8;
#pragma unroll
  for (int t = 0; t < 2; ++t) {
    floatx8 ac = t ? acc1 : acc0;
    int cout = nt * 32 + t * 16 + nl;    // D layout: lane%16 = N (channel)
    bool cok = cout < P.Cout;
    float ls = 0.f, lq = 0.f;
#pragma unroll
    for (int rI = 0; rI < 8; ++rI) {     // D layout: voxel = base + half*8 + r
      long mv = mBase + rI;
      if (cok && mv < nvox) {
        float vv = ac[rI];
        P.out[((long)n * P.Cout + cout) * nvox + mv] = vv;
        ls += vv; lq += vv * vv;
      }
    }
    if (cok) { atomicAdd(&sSum[t * 16 + nl], ls); atomicAdd(&sSq[t * 16 + nl], lq); }
  }
  __syncthreads();
  if (tid < 32 && nt * 32 + tid < P.Cout) {
    long sidx = ((long)n * P.Cout + nt * 32 + tid) * 2;
    atomicAdd(&P.stats[sidx + 0], sSum[tid]);
    atomicAdd(&P.stats[sidx + 1], sSq[tid]);
  }
}

// fp32 weights -> packed bf16 in WMMA *fragment order*: element i is exactly
// the bf16 that lane (i>>4)&31 needs at register slot i&15, so the device
// side reads whole fragments with b128 LDS loads. k = half*16 + pos.
__global__ __launch_bounds__(256) void tcn_pack_weights_bf16(
    const float* w, __bf16* dst, int Cin, int Cout,
    int KH, int KW, int KD, int KT, int transpose,
    int nCoTt, int ctb, int nCC, int ccb, long total) {
  long i = (long)blockIdx.x * 256 + threadIdx.x;
  if (i >= total) return;
  int pos  = (int)(i & 15);
  int lane = (int)((i >> 4) & 31);
  long r = i >> 9;
  int ct = (int)(r & (nCoTt - 1)); r >>= ctb;
  int cc = (int)(r & (nCC - 1));
  int kpos = (int)(r >> ccb);
  int k  = (lane >> 4) * 16 + pos;
  int ci = cc * 32 + k;
  int co = ct * 16 + (lane & 15);
  float v = 0.f;
  if (ci < Cin && co < Cout) {
    int kt = kpos % KT; int q = kpos / KT;
    int kd = q % KD; q /= KD;
    int kw = q % KW;
    int kh = q / KW;
    long idx;
    if (transpose) {  // (in,out,...) layout, spatially flipped taps
      kh = KH - 1 - kh; kw = KW - 1 - kw; kd = KD - 1 - kd; kt = KT - 1 - kt;
      idx = (((long)ci * Cout + co) * KH + kh) * KW + kw;
    } else {
      idx = (((long)co * Cin + ci) * KH + kh) * KW + kw;
    }
    idx = (idx * KD + kd) * KT + kt;
    v = w[idx];
  }
  dst[i] = (__bf16)v;
}

__global__ void tcn_finalize_stats(float* stats, int NC, float invCount) {
  int i = blockIdx.x * 256 + threadIdx.x;
  if (i >= NC) return;
  float s = stats[2 * i], q = stats[2 * i + 1];
  float m = s * invCount;
  float var = fmaxf(q * invCount - m * m, 0.f);
  stats[2 * i] = m;
  stats[2 * i + 1] = rsqrtf(var + 1e-5f);
}

// (x-mean)*rsqrt(var+eps) + ELU (+ skip), channel-first fp32 -> channel-last bf16.
__global__ void tcn_norm_elu_bf16(const float* conv, const float* stats,
                                  __bf16* dst, const __bf16* skip,
                                  int C, int logC, long nvox, int lognv,
                                  long total) {
  long i = (long)blockIdx.x * 256 + threadIdx.x;
  if (i >= total) return;
  int c = (int)(i & (C - 1));
  long r = i >> logC;
  long mv = r & (nvox - 1);
  int n = (int)(r >> lognv);
  long sidx = ((long)n * C + c) * 2;
  float v = (conv[((long)n * C + c) * nvox + mv] - stats[sidx]) * stats[sidx + 1];
  v = v > 0.f ? v : (__expf(v) - 1.f);
  if (skip) v += (float)skip[i];
  dst[i] = (__bf16)v;
}

// Pruning head: channel-last makes the 1x1 conv a contiguous dot per voxel.
__global__ void tcn_prune_bf16(__bf16* y, const float* wp, const float* bp,
                               int C, long total, float alpha) {
  long i = (long)blockIdx.x * 256 + threadIdx.x;
  if (i >= total) return;                     // i over N*nvox voxels
  __bf16* yv = y + i * (long)C;
  float logit = bp[0];
  for (int c = 0; c < C; ++c) logit += wp[c] * (float)yv[c];
  if (!(logit > alpha))
    for (int c = 0; c < C; ++c) yv[c] = (__bf16)0.0f;
}

__global__ void tcn_sigmoid_out(const float* conv, const float* bias,
                                float* out, int lognv, long total) {
  long i = (long)blockIdx.x * 256 + threadIdx.x;
  if (i >= total) return;
  int c = (int)((i >> lognv) % 3u);
  float v = conv[i] + bias[c];
  out[i] = 1.f / (1.f + __expf(-v));
}

// Input x: channel-first fp32 -> channel-last bf16.
__global__ void tcn_in_to_bf16(const float* src, __bf16* dst, long nvox,
                               int lognv, long total) {
  long i = (long)blockIdx.x * 256 + threadIdx.x;
  if (i >= total) return;
  long q = i / 3;
  int c = (int)(i - q * 3);
  long mv = q & (nvox - 1);
  int n = (int)(q >> lognv);
  dst[i] = (__bf16)src[((long)n * 3 + c) * nvox + mv];
}

// ---------------------------------------------------------------------------
// Host orchestration
// ---------------------------------------------------------------------------
struct LayerDims {
  int Cin, Cout;
  int IH, IW, ID, IT, OH, OW, OD, OT;
  int KH, KW, KD, KT;
  int sh, ph, pt;
  int transpose;
};

static inline int ilog2i(int x) { int l = 0; while ((1 << l) < x) ++l; return l; }

static void run_conv(const LayerDims& L, const float* w, const __bf16* actIn,
                     __bf16* wpk, float* convout, float* stats,
                     int N, hipStream_t s) {
  const int nCC = (L.Cin + 31) >> 5;
  const int nCoTt = 2 * ((L.Cout + 31) >> 5);
  const int kvol = L.KH * L.KW * L.KD * L.KT;
  const long total = (long)kvol * nCC * nCoTt * 512;
  tcn_pack_weights_bf16<<<(unsigned)((total + 255) / 256), 256, 0, s>>>(
      w, wpk, L.Cin, L.Cout, L.KH, L.KW, L.KD, L.KT, L.transpose,
      nCoTt, ilog2i(nCoTt), nCC, ilog2i(nCC), total);
  (void)hipMemsetAsync(stats, 0, (size_t)N * L.Cout * 2 * sizeof(float), s);
  ConvParams P;
  P.act = actIn; P.wpk = wpk; P.out = convout; P.stats = stats;
  P.Cin = L.Cin; P.Cout = L.Cout;
  P.IH = L.IH; P.IW = L.IW; P.ID = L.ID; P.IT = L.IT;
  P.OH = L.OH; P.OW = L.OW; P.OD = L.OD; P.OT = L.OT;
  P.OTb = ilog2i(L.OT); P.ODb = ilog2i(L.OD); P.OWb = ilog2i(L.OW);
  P.KH = L.KH; P.KW = L.KW; P.KD = L.KD; P.KT = L.KT;
  P.sh = L.sh; P.ph = L.ph; P.pt = L.pt;
  P.transpose = L.transpose;
  P.nCC = nCC; P.nCoTt = nCoTt;
  const long nvox = (long)L.OH * L.OW * L.OD * L.OT;
  dim3 grid((unsigned)((nvox + 127) / 128), (unsigned)(nCoTt / 2), (unsigned)N);
  tcn_conv3d_wmma_bf16<<<grid, 256, 0, s>>>(P);
  const int NC = N * L.Cout;
  tcn_finalize_stats<<<(NC + 255) / 256, 256, 0, s>>>(stats, NC, 1.0f / (float)nvox);
}

extern "C" void kernel_launch(void* const* d_in, const int* in_sizes, int n_in,
                              void* d_out, int out_size, void* d_ws, size_t ws_size,
                              hipStream_t stream) {
  (void)in_sizes; (void)n_in; (void)out_size; (void)ws_size;
  const int N = 2, T = 2;
  static const int LH[6] = {128, 64, 32, 16, 8, 4};
  static const int LD[6] = {32, 16, 8, 4, 2, 1};
  static const int ENCC[6] = {8, 16, 32, 64, 128, 256};

  char* ws = (char*)d_ws;
  size_t off = 0;
  auto arena = [&](size_t bytes) -> void* {
    off = (off + 255) & ~(size_t)255;
    void* p = ws + off;
    off += bytes + 4096;    // slack for benign overshoot reads
    return p;
  };
  const size_t nv0 = (size_t)LH[0] * LH[0] * LD[0] * T;   // 1,048,576 voxels
  __bf16* wpk     = (__bf16*)arena((size_t)16 << 20);
  float*  convout = (float*)arena((size_t)N * 8 * nv0 * sizeof(float));
  float*  stats   = (float*)arena(4096);
  __bf16* actA    = (__bf16*)arena((size_t)N * 8 * nv0 * sizeof(__bf16));
  __bf16* actB    = (__bf16*)arena((size_t)N * 8 * nv0 * sizeof(__bf16));
  __bf16* skipBuf[5];
  for (int l = 0; l < 5; ++l) {
    size_t nv = (size_t)LH[l] * LH[l] * LD[l] * T;
    skipBuf[l] = (__bf16*)arena((size_t)N * ENCC[l] * nv * sizeof(__bf16));
  }

  auto mkL = [&](int Cin, int Cout, int lin, int lout,
                 int KH, int KW, int KD, int KT,
                 int sh, int ph, int pt, int tr) -> LayerDims {
    LayerDims L;
    L.Cin = Cin; L.Cout = Cout;
    L.IH = LH[lin]; L.IW = LH[lin]; L.ID = LD[lin]; L.IT = T;
    L.OH = LH[lout]; L.OW = LH[lout]; L.OD = LD[lout]; L.OT = T;
    L.KH = KH; L.KW = KW; L.KD = KD; L.KT = KT;
    L.sh = sh; L.ph = ph; L.pt = pt;
    L.transpose = tr;
    return L;
  };
  auto runConvNorm = [&](const LayerDims& L, const float* w, const __bf16* in,
                         __bf16* dst, const __bf16* skip) {
    run_conv(L, w, in, wpk, convout, stats, N, stream);
    long nvox = (long)L.OH * L.OW * L.OD * L.OT;
    long tot = (long)N * L.Cout * nvox;
    tcn_norm_elu_bf16<<<(unsigned)((tot + 255) / 256), 256, 0, stream>>>(
        convout, stats, dst, skip, L.Cout, ilog2i(L.Cout), nvox,
        ilog2i((int)nvox) , tot);
  };

  // Input x (fp32, channel-first) -> channel-last bf16.
  const float* x = (const float*)d_in[0];
  long xtot = (long)N * 3 * nv0;
  tcn_in_to_bf16<<<(unsigned)((xtot + 255) / 256), 256, 0, stream>>>(
      x, actA, (long)nv0, ilog2i((int)nv0), xtot);

  // ---- Encoder ----
  runConvNorm(mkL(3, 8, 0, 0, 3, 3, 3, 2, 1, 1, 0, 0), (const float*)d_in[1],
              actA, skipBuf[0], nullptr);
  const __bf16* enc = skipBuf[0];
  for (int i = 0; i < 5; ++i) {
    int ci = ENCC[i], co = ENCC[i + 1];
    // stride-2 SAME (k=3, even extent): pad_lo = 0
    runConvNorm(mkL(ci, co, i, i + 1, 3, 3, 3, 2, 2, 0, 0, 0),
                (const float*)d_in[2 + 2 * i], enc, actA, nullptr);
    __bf16* dst = (i < 4) ? skipBuf[i + 1] : actB;
    runConvNorm(mkL(co, co, i + 1, i + 1, 3, 3, 3, 2, 1, 1, 0, 0),
                (const float*)d_in[3 + 2 * i], actA, dst, nullptr);
    enc = dst;
  }

  // ---- Decoder ----
  __bf16* cur = actB;   // e162: 256ch @ level 5
  __bf16* other = actA;
  for (int i = 0; i < 5; ++i) {
    int ci = ENCC[5 - i], co = ENCC[4 - i];
    int lin = 5 - i, lout = 4 - i;
    const float* wt  = (const float*)d_in[12 + 5 * i + 0];
    const float* wc1 = (const float*)d_in[12 + 5 * i + 1];
    const float* wc2 = (const float*)d_in[12 + 5 * i + 2];
    const float* wp  = (const float*)d_in[12 + 5 * i + 3];
    const float* bp  = (const float*)d_in[12 + 5 * i + 4];
    // generative transpose conv x2 spatial (gather form, pad 1)
    runConvNorm(mkL(ci, co, lin, lout, 3, 3, 3, 1, 2, 1, 0, 1), wt, cur, other, nullptr);
    { __bf16* t = cur; cur = other; other = t; }
    // conv + IN + ELU, then skip add
    runConvNorm(mkL(co, co, lout, lout, 3, 3, 3, 1, 1, 1, 0, 0), wc1, cur, other,
                skipBuf[4 - i]);
    { __bf16* t = cur; cur = other; other = t; }
    // dec_block_2: 2x2x2x2 conv (pad_lo 0) + IN + ELU
    runConvNorm(mkL(co, co, lout, lout, 2, 2, 2, 2, 1, 0, 0, 0), wc2, cur, other, nullptr);
    { __bf16* t = cur; cur = other; other = t; }
    // pruning head
    long nvoxl = (long)LH[lout] * LH[lout] * LD[lout] * T;
    long ptot = (long)N * nvoxl;
    tcn_prune_bf16<<<(unsigned)((ptot + 255) / 256), 256, 0, stream>>>(
        cur, wp, bp, co, ptot, 0.0f);
  }

  // ---- Centroid head: 8->3 conv k(3,3,3,1) pad 1 + bias + sigmoid ----
  {
    LayerDims L = mkL(8, 3, 0, 0, 3, 3, 3, 1, 1, 1, 0, 0);
    run_conv(L, (const float*)d_in[37], cur, wpk, convout, stats, N, stream);
    long tot = (long)N * 3 * (long)nv0;
    tcn_sigmoid_out<<<(unsigned)((tot + 255) / 256), 256, 0, stream>>>(
        convout, (const float*)d_in[38], (float*)d_out, ilog2i((int)nv0), tot);
  }
}
